// rNN_16312285790591
// MI455X (gfx1250) — matrix operand order repaired and checked by reading, not compiled, not run on hardware
//
#include <hip/hip_runtime.h>

// LSTM (B=4096, T=512, F=32, H=60) + dense head, bf16-WMMA persistent kernel.
typedef __attribute__((ext_vector_type(16))) __bf16 v16bf;
typedef __attribute__((ext_vector_type(8)))  __bf16 v8bf;
typedef __attribute__((ext_vector_type(8)))  float  v8f;

#define TT   512
#define FF   32
#define HH   60
#define NG   240          // 4*H
#define KP   96           // K = F+H = 92, padded to 96 (3 chunks of 32)
#define AROW 104          // A row stride in halves (padded: 52 dwords, conflict-free)
#define GS   244          // gates row stride in floats (conflict-free scatter)
#define MT   16           // batch rows per block
#define NTHREADS 128      // 4 waves

__device__ __forceinline__ float sigf(float v) {
    return __builtin_amdgcn_rcpf(1.0f + __expf(-v));
}
__device__ __forceinline__ float tanhfast(float v) {
    return 2.0f * __builtin_amdgcn_rcpf(1.0f + __expf(-2.0f * v)) - 1.0f;
}

__global__ __launch_bounds__(NTHREADS, 1)
void lstm_wmma_kernel(const float* __restrict__ x,        // [B,T,F]
                      const float* __restrict__ kernel,   // [F+H, 4H]
                      const float* __restrict__ bias,     // [4H]
                      const float* __restrict__ dense_w,  // [H,1]
                      const float* __restrict__ dense_b,  // [1]
                      float* __restrict__ out)            // [B,1]
{
    __shared__ __bf16 WtLds[NG * KP];    // 46080 B : Wt[n][k] = kernel[k][n], bf16
    __shared__ __bf16 ALds [MT * AROW];  //  3328 B : A[m][k] = [x_t | h_{t-1}]
    __shared__ float  GLds [MT * GS];    // 15616 B : gate pre-activations / partials

    const int tid  = threadIdx.x;
    const int lane = tid & 31;
    // Wave id as a *scalar* (SGPR) value: tile guards below must be uniform
    // branches so EXEC stays all-ones around every WMMA (ISA requirement).
    const int wave = __builtin_amdgcn_readfirstlane(tid >> 5);
    const int nl   = lane & 15;
    const int b0   = blockIdx.x * MT;

    // -------- prologue: stage transposed bf16 weights into LDS --------
    for (int idx = tid; idx < NG * KP; idx += NTHREADS) {
        int n = idx / KP, k = idx - n * KP;
        float v = (k < FF + HH) ? kernel[(size_t)k * NG + n] : 0.0f;
        WtLds[idx] = (__bf16)v;
    }
    for (int idx = tid; idx < MT * AROW; idx += NTHREADS)
        ALds[idx] = (__bf16)0.0f;   // h_0 = 0, pad columns = 0
    __syncthreads();

    // -------- B fragments + bias: resident in registers for all T steps --------
    v16bf bfr[4][3];
    float bs[4];
    #pragma unroll
    for (int i = 0; i < 4; ++i) {
        int nt = wave * 4 + i;                    // N-tile id (15 valid tiles), uniform
        int n  = (nt < 15) ? (nt * 16 + nl) : nl; // clamp for the one invalid tile
        bs[i]  = (nt < 15) ? bias[n] : 0.0f;
        #pragma unroll
        for (int kc = 0; kc < 3; ++kc) {
            int kb = kc * 32 + ((lane < 16) ? 0 : 16);  // B layout: lanes 0-15 K lo, 16-31 K hi
            bfr[i][kc] = *(const v16bf*)&WtLds[n * KP + kb];
        }
    }

    // -------- per-thread elementwise ownership: 8 (b,h) cells --------
    const int eb = tid >> 3;         // batch row within tile (also x-loader row)
    const int hg = (tid & 7) * 8;    // h base (h in [hg, hg+8))
    const int fq = (tid & 7) * 4;    // x feature quad for loads
    float c_[8], hreg[8];
    #pragma unroll
    for (int e = 0; e < 8; ++e) { c_[e] = 0.0f; hreg[e] = 0.0f; }

    // x_0 into A
    {
        const float4 x0 = *(const float4*)&x[(size_t)(b0 + eb) * TT * FF + fq];
        ALds[eb * AROW + fq + 0] = (__bf16)x0.x;
        ALds[eb * AROW + fq + 1] = (__bf16)x0.y;
        ALds[eb * AROW + fq + 2] = (__bf16)x0.z;
        ALds[eb * AROW + fq + 3] = (__bf16)x0.w;
    }
    __syncthreads();

    // -------- time loop --------
    for (int t = 0; t < TT; ++t) {
        // stream next x tile into registers first (VMEM latency overlaps WMMA);
        // prefetch two steps ahead (global_prefetch_b8)
        float4 xn = make_float4(0.f, 0.f, 0.f, 0.f);
        if (t + 1 < TT)
            xn = *(const float4*)&x[(size_t)(b0 + eb) * TT * FF + (size_t)(t + 1) * FF + fq];
        if (t + 2 < TT)
            __builtin_prefetch(&x[(size_t)(b0 + eb) * TT * FF + (size_t)(t + 2) * FF + fq], 0, 3);

        // A fragments (contiguous K order matching B: lanes 0-15 K lo 16, 16-31 K hi 16)
        v16bf af[3];
        #pragma unroll
        for (int kc = 0; kc < 3; ++kc) {
            int k1 = kc * 32 + ((lane < 16) ? 0 : 16);
            v8bf lo = *(const v8bf*)&ALds[nl * AROW + k1];
            v8bf hi = *(const v8bf*)&ALds[nl * AROW + k1 + 8];
            af[kc] = __builtin_shufflevector(lo, hi, 0,1,2,3,4,5,6,7,8,9,10,11,12,13,14,15);
        }

        // gates = A * Wt + bias  (guards are SGPR-uniform -> scalar branches,
        // EXEC remains all-ones around every WMMA)
        #pragma unroll
        for (int i = 0; i < 4; ++i) {
            int nt = wave * 4 + i;
            if (nt < 15) {
                v8f acc;
                #pragma unroll
                for (int r = 0; r < 8; ++r) acc[r] = bs[i];
                #pragma unroll
                for (int kc = 0; kc < 3; ++kc)
                    acc = __builtin_amdgcn_wmma_f32_16x16x32_bf16(
                              false, af[kc], false, bfr[i][kc],
                              (short)0, acc, false, false);
                // scatter per C/D layout: VGPR r -> M = r (+8 for upper lanes), N = nt*16+nl
                int gn = nt * 16 + nl;
                int gm = (lane < 16) ? 0 : 8;
                #pragma unroll
                for (int r = 0; r < 8; ++r)
                    GLds[(gm + r) * GS + gn] = acc[r];
            }
        }
        __syncthreads();

        // elementwise LSTM cell update (gate order: i, j, f, o)
        #pragma unroll
        for (int e = 0; e < 8; ++e) {
            int h = hg + e;
            float nh = 0.0f;
            if (h < HH) {
                float gi = GLds[eb * GS + h];
                float gj = GLds[eb * GS + 60 + h];
                float gf = GLds[eb * GS + 120 + h];
                float go = GLds[eb * GS + 180 + h];
                float nc = c_[e] * sigf(gf + 1.0f) + sigf(gi) * tanhfast(gj);
                c_[e] = nc;
                nh = tanhfast(nc) * sigf(go);
            }
            hreg[e] = (h < HH) ? nh : 0.0f;
            ALds[eb * AROW + FF + h] = (__bf16)nh;   // h (and zero K-pad) for next step
        }
        // commit next x tile into A
        if (t + 1 < TT) {
            ALds[eb * AROW + fq + 0] = (__bf16)xn.x;
            ALds[eb * AROW + fq + 1] = (__bf16)xn.y;
            ALds[eb * AROW + fq + 2] = (__bf16)xn.z;
            ALds[eb * AROW + fq + 3] = (__bf16)xn.w;
        }
        __syncthreads();
    }

    // -------- dense head: out[b] = h_T . dense_w + dense_b --------
    float p = 0.0f;
    #pragma unroll
    for (int e = 0; e < 8; ++e) {
        int h = hg + e;
        if (h < HH) p += hreg[e] * dense_w[h];
    }
    GLds[tid] = p;              // reuse gate buffer for 128 partials
    __syncthreads();
    if (tid < MT) {
        float s = dense_b[0];
        #pragma unroll
        for (int g = 0; g < 8; ++g) s += GLds[tid * 8 + g];
        out[b0 + tid] = s;
    }
}

extern "C" void kernel_launch(void* const* d_in, const int* in_sizes, int n_in,
                              void* d_out, int out_size, void* d_ws, size_t ws_size,
                              hipStream_t stream) {
    const float* x       = (const float*)d_in[0];
    const float* kernel  = (const float*)d_in[1];
    const float* bias    = (const float*)d_in[2];
    const float* dense_w = (const float*)d_in[3];
    const float* dense_b = (const float*)d_in[4];
    float* out = (float*)d_out;

    const int B = 4096;
    dim3 grid(B / MT), block(NTHREADS);
    lstm_wmma_kernel<<<grid, block, 0, stream>>>(x, kernel, bias, dense_w, dense_b, out);
}
